// VideoDepthAnything_83227876262125
// MI455X (gfx1250) — matrix operand order repaired and checked by reading, not compiled, not run on hardware
//
#include <hip/hip_runtime.h>
#include <hip/hip_bf16.h>

typedef __attribute__((ext_vector_type(16))) __bf16 v16bf;
typedef __attribute__((ext_vector_type(8)))  __bf16 v8bf;
typedef __attribute__((ext_vector_type(4)))  __bf16 v4bf;
typedef __attribute__((ext_vector_type(8)))  float  v8f;

#define NUM_HEADS 8
#define N_POINTS  8
#define HEAD_DIM  128

// ---------------- bf16 WMMA GEMM: C = A(MxK,f32) * B(KxN,f32) + bias ----------------
// Tile: 128(M) x 128(N), BK=32, LDS ping-pong double buffered. 256 threads = 8 waves;
// wave w computes rows [m0+16w, +16) x 128 cols via 8x v_wmma_f32_16x16x32_bf16.
#define BM 128
#define BN 128
#define BK 32
#define LDS_STR 56   // bf16 elems; 112B rows: 16B-aligned chunks, 28-bank step (conflict-free)

static __device__ __forceinline__ unsigned pack_bf16x2(float lo, float hi) {
    unsigned short a = __builtin_bit_cast(unsigned short, (__bf16)lo);
    unsigned short b = __builtin_bit_cast(unsigned short, (__bf16)hi);
    return (unsigned)a | ((unsigned)b << 16);
}

static __device__ __forceinline__ v16bf load_frag(const __bf16* s, int row, int kofs) {
    const v8bf lo = *(const v8bf*)(s + row * LDS_STR + kofs);
    const v8bf hi = *(const v8bf*)(s + row * LDS_STR + kofs + 8);
    return __builtin_shufflevector(lo, hi, 0,1,2,3,4,5,6,7,8,9,10,11,12,13,14,15);
}

struct StageRegs { float4 av[4]; float4 bv0[2]; float4 bv1[2]; };

__global__ __launch_bounds__(256)
void gemm_bf16_wmma(const float* __restrict__ A, int lda,
                    const float* __restrict__ Bm, int ldb,
                    const float* __restrict__ bias,
                    float* __restrict__ C, int ldc,
                    int M, int N, int K)
{
    __shared__ __bf16 sA[2][BM * LDS_STR];   // 2 x 14 KB ping-pong
    __shared__ __bf16 sB[2][BN * LDS_STR];   // 2 x 14 KB, transposed [n][k]

    const int tid  = threadIdx.x;
    const int lane = tid & 31;
    const int wave = tid >> 5;

    const int m0 = blockIdx.x * BM;
    const int n0 = blockIdx.y * BN;
    const bool full = (m0 + BM) <= M;     // uniform: only last M-block is partial

    // staging coordinates
    const int a_row0 = tid >> 3;          // 0..31 (A: 4 row-groups of 32)
    const int a_col  = (tid & 7) * 4;     // 0..28
    const int b_kp   = (tid & 15) * 2;    // K pair base 0..30
    const int b_cg   = tid >> 4;          // 0..15 (B: 2 col-group passes of 16)

    // ---- staging helpers (issue loads / convert+store) ----
    auto stage_load = [&](int k0, StageRegs& rg) {
        if (full) {
            #pragma unroll
            for (int r = 0; r < 4; ++r)
                rg.av[r] = *(const float4*)(A + (size_t)(m0 + a_row0 + r * 32) * lda + k0 + a_col);
        } else {
            #pragma unroll
            for (int r = 0; r < 4; ++r) {
                rg.av[r] = make_float4(0.f, 0.f, 0.f, 0.f);
                if (m0 + a_row0 + r * 32 < M)
                    rg.av[r] = *(const float4*)(A + (size_t)(m0 + a_row0 + r * 32) * lda + k0 + a_col);
            }
        }
        #pragma unroll
        for (int it = 0; it < 2; ++it) {
            const float* src = Bm + (size_t)(k0 + b_kp) * ldb + n0 + (b_cg + it * 16) * 4;
            rg.bv0[it] = *(const float4*)(src);
            rg.bv1[it] = *(const float4*)(src + ldb);
        }
    };
    auto stage_store = [&](__bf16* dA, __bf16* dB, const StageRegs& rg) {
        #pragma unroll
        for (int r = 0; r < 4; ++r)
            *(v4bf*)(dA + (a_row0 + r * 32) * LDS_STR + a_col) =
                (v4bf){(__bf16)rg.av[r].x, (__bf16)rg.av[r].y,
                       (__bf16)rg.av[r].z, (__bf16)rg.av[r].w};
        #pragma unroll
        for (int it = 0; it < 2; ++it) {
            const int n = (b_cg + it * 16) * 4;
            *(unsigned*)(dB + (n + 0) * LDS_STR + b_kp) = pack_bf16x2(rg.bv0[it].x, rg.bv1[it].x);
            *(unsigned*)(dB + (n + 1) * LDS_STR + b_kp) = pack_bf16x2(rg.bv0[it].y, rg.bv1[it].y);
            *(unsigned*)(dB + (n + 2) * LDS_STR + b_kp) = pack_bf16x2(rg.bv0[it].z, rg.bv1[it].z);
            *(unsigned*)(dB + (n + 3) * LDS_STR + b_kp) = pack_bf16x2(rg.bv0[it].w, rg.bv1[it].w);
        }
    };

    v8f acc[8] = {};

    // prologue: stage tile 0 into buffer 0
    {
        StageRegs rg;
        stage_load(0, rg);
        stage_store(sA[0], sB[0], rg);
    }
    __syncthreads();

    const int kb    = (lane >> 4) * 16;      // A frag K offset (0/16)
    const int arow  = wave * 16 + (lane & 15);
    const int kb2   = (lane >> 4) * 16;      // B frag K offset (0/16)
    const int bcol0 = lane & 15;

    int cur = 0;
    for (int k0 = 0; k0 < K; k0 += BK, cur ^= 1) {
        const bool have_next = (k0 + BK) < K;

        // 1) issue global loads for the NEXT tile (in flight during the WMMA block)
        StageRegs rg;
        if (have_next) stage_load(k0 + BK, rg);
        if (k0 + 2 * BK < K) {
            const int pr = full ? (m0 + a_row0) : min(m0 + a_row0, M - 1);
            __builtin_prefetch((const void*)(A + (size_t)pr * lda + k0 + 2 * BK), 0, 0);
            __builtin_prefetch((const void*)(Bm + (size_t)(k0 + 2 * BK + (tid & 31)) * ldb + n0), 0, 0);
        }

        // 2) compute from the CURRENT buffer: preload all fragments, dense WMMA block.
        const __bf16* sAc = sA[cur];
        const __bf16* sBc = sB[cur];
        const v8bf a_lo = *(const v8bf*)(sAc + arow * LDS_STR + kb);
        const v8bf a_hi = *(const v8bf*)(sAc + arow * LDS_STR + kb + 8);
        const v16bf a_frag = __builtin_shufflevector(a_lo, a_hi,
            0,1,2,3,4,5,6,7,8,9,10,11,12,13,14,15);

        v16bf bfr[8];
        #pragma unroll
        for (int j = 0; j < 8; ++j)
            bfr[j] = load_frag(sBc, j * 16 + bcol0, kb2);
        #pragma unroll
        for (int j = 0; j < 8; ++j)
            acc[j] = __builtin_amdgcn_wmma_f32_16x16x32_bf16(
                false, a_frag, false, bfr[j], (short)0, acc[j], false, false);

        // 3) convert + store next tile into the OTHER buffer; one barrier per step
        if (have_next) stage_store(sA[cur ^ 1], sB[cur ^ 1], rg);
        __syncthreads();
    }

    // Epilogue: C/D layout -> VGPR r: lanes0-15 M=r, lanes16-31 M=r+8; N=lane&15.
    const int mo  = (lane >> 4) * 8;
    const int col = lane & 15;
    if (full) {
        #pragma unroll
        for (int j = 0; j < 8; ++j) {
            const int gn = n0 + j * 16 + col;
            const float bv = bias[gn];
            #pragma unroll
            for (int r = 0; r < 8; ++r) {
                const int gm = m0 + wave * 16 + mo + r;
                C[(size_t)gm * ldc + gn] = acc[j][r] + bv;
            }
        }
    } else {
        #pragma unroll
        for (int j = 0; j < 8; ++j) {
            const int gn = n0 + j * 16 + col;
            const float bv = bias[gn];
            #pragma unroll
            for (int r = 0; r < 8; ++r) {
                const int gm = m0 + wave * 16 + mo + r;
                if (gm < M) C[(size_t)gm * ldc + gn] = acc[j][r] + bv;
            }
        }
    }
}

// ---------------- deformable sampling + 8-point softmax attention ----------------
// One wave32 per (b, n, h). Lane owns channels [4*lane, 4*lane+4) of HEAD_DIM=128.
__global__ __launch_bounds__(256)
void deform_attn_kernel(const float* __restrict__ q,    // (B,N,C)
                        const float* __restrict__ k,    // (B,N,C)
                        const float* __restrict__ off,  // (B,N,H*P*2)
                        float* __restrict__ out,        // (B,N,C)
                        int Bn, int Np, int Hh, int Ww)
{
    const int C = NUM_HEADS * HEAD_DIM;
    const int wg   = (blockIdx.x * blockDim.x + threadIdx.x) >> 5;
    const int lane = threadIdx.x & 31;
    const int total = Bn * Np * NUM_HEADS;
    if (wg >= total) return;

    const int h = wg % NUM_HEADS;
    const int n = (wg / NUM_HEADS) % Np;
    const int b = wg / (NUM_HEADS * Np);

    const int iyn = n / Ww, ixn = n % Ww;
    const float yy = -1.0f + 2.0f * (float)iyn / (float)(Hh - 1);
    const float xx = -1.0f + 2.0f * (float)ixn / (float)(Ww - 1);
    const float scale = rsqrtf((float)HEAD_DIM);

    const size_t baseqk = ((size_t)b * Np + n) * C + h * HEAD_DIM;
    const float4 qv = *(const float4*)(q + baseqk + lane * 4);
    const float* offp = off + ((size_t)b * Np + n) * (NUM_HEADS * N_POINTS * 2)
                            + h * (N_POINTS * 2);
    const float* kbp = k + (size_t)b * Np * C + h * HEAD_DIM + lane * 4;

    float4 sk[N_POINTS];
    float  sc[N_POINTS];

    #pragma unroll
    for (int p = 0; p < N_POINTS; ++p) {
        const float gx = yy + offp[p * 2 + 0];   // reference pairs off[...,0] with yy
        const float gy = xx + offp[p * 2 + 1];
        float ix = (gx + 1.0f) * 0.5f * (float)(Ww - 1);
        float iy = (gy + 1.0f) * 0.5f * (float)(Hh - 1);
        ix = fminf(fmaxf(ix, 0.0f), (float)(Ww - 1));
        iy = fminf(fmaxf(iy, 0.0f), (float)(Hh - 1));
        const float x0 = floorf(ix), y0 = floorf(iy);
        const float wx = ix - x0,   wy = iy - y0;
        const int x0i = (int)x0, y0i = (int)y0;
        const int x1i = min(x0i + 1, Ww - 1);
        const int y1i = min(y0i + 1, Hh - 1);

        const float4 k00 = *(const float4*)(kbp + (size_t)(y0i * Ww + x0i) * C);
        const float4 k01 = *(const float4*)(kbp + (size_t)(y0i * Ww + x1i) * C);
        const float4 k10 = *(const float4*)(kbp + (size_t)(y1i * Ww + x0i) * C);
        const float4 k11 = *(const float4*)(kbp + (size_t)(y1i * Ww + x1i) * C);
        const float w00 = (1.f - wy) * (1.f - wx), w01 = (1.f - wy) * wx;
        const float w10 = wy * (1.f - wx),         w11 = wy * wx;

        float4 s;
        s.x = k00.x * w00 + k01.x * w01 + k10.x * w10 + k11.x * w11;
        s.y = k00.y * w00 + k01.y * w01 + k10.y * w10 + k11.y * w11;
        s.z = k00.z * w00 + k01.z * w01 + k10.z * w10 + k11.z * w11;
        s.w = k00.w * w00 + k01.w * w01 + k10.w * w10 + k11.w * w11;
        sk[p] = s;

        float d = qv.x * s.x + qv.y * s.y + qv.z * s.z + qv.w * s.w;
        #pragma unroll
        for (int m = 16; m >= 1; m >>= 1) d += __shfl_xor(d, m, 32);
        sc[p] = d * scale;
    }

    float mx = sc[0];
    #pragma unroll
    for (int p = 1; p < N_POINTS; ++p) mx = fmaxf(mx, sc[p]);
    float ex[N_POINTS], se = 0.f;
    #pragma unroll
    for (int p = 0; p < N_POINTS; ++p) { ex[p] = __expf(sc[p] - mx); se += ex[p]; }
    const float inv = 1.0f / se;

    float4 o = make_float4(0.f, 0.f, 0.f, 0.f);
    #pragma unroll
    for (int p = 0; p < N_POINTS; ++p) {
        const float a = ex[p] * inv;
        o.x += a * sk[p].x; o.y += a * sk[p].y;
        o.z += a * sk[p].z; o.w += a * sk[p].w;
    }
    *(float4*)(out + baseqk + lane * 4) = o;
}

// ---------------- host-side launch ----------------
extern "C" void kernel_launch(void* const* d_in, const int* in_sizes, int n_in,
                              void* d_out, int out_size, void* d_ws, size_t ws_size,
                              hipStream_t stream)
{
    const float* query = (const float*)d_in[0];
    const float* ref   = (const float*)d_in[1];
    const float* Wq    = (const float*)d_in[2];
    const float* bq    = (const float*)d_in[3];
    const float* Wkv   = (const float*)d_in[4];
    const float* bkv   = (const float*)d_in[5];
    const float* Woff  = (const float*)d_in[6];
    const float* boff  = (const float*)d_in[7];
    const float* Wout  = (const float*)d_in[8];
    const float* bout  = (const float*)d_in[9];
    float* out = (float*)d_out;

    const int C    = in_sizes[3];            // 1024
    const int OFFC = in_sizes[7];            // NUM_HEADS*N_POINTS*2 = 128
    const int Hh = 37, Ww = 37;
    const int Np = Hh * Ww;                  // 1369
    const int Mtot = in_sizes[0] / C;        // B*N = 10952
    const int Bn = Mtot / Np;                // 8

    // workspace carve-out (f32), 256B-aligned
    char* ws = (char*)d_ws;
    size_t o = 0;
    auto carve = [&](size_t bytes) { void* p = ws + o; o = (o + bytes + 255) & ~(size_t)255; return p; };
    float* q_buf    = (float*)carve((size_t)Mtot * C * sizeof(float));
    float* k_buf    = (float*)carve((size_t)Mtot * C * sizeof(float));
    float* off_buf  = (float*)carve((size_t)Mtot * OFFC * sizeof(float));
    float* attn_buf = (float*)carve((size_t)Mtot * C * sizeof(float));
    (void)ws_size;

    const dim3 blk(256);
    const int gM = (Mtot + BM - 1) / BM;

    // q = query @ Wq + bq
    gemm_bf16_wmma<<<dim3(gM, C / BN), blk, 0, stream>>>(
        query, C, Wq, C, bq, q_buf, C, Mtot, C, C);
    // k = ref @ Wkv[:, :C] + bkv[:C]   (v half of kv is dead code in the reference)
    gemm_bf16_wmma<<<dim3(gM, C / BN), blk, 0, stream>>>(
        ref, C, Wkv, 2 * C, bkv, k_buf, C, Mtot, C, C);
    // off = query @ Woff + boff
    gemm_bf16_wmma<<<dim3(gM, OFFC / BN), blk, 0, stream>>>(
        query, C, Woff, OFFC, boff, off_buf, OFFC, Mtot, OFFC, C);

    // deformable sampling + attention: one wave per (b,n,h)
    const int totalWaves = Bn * Np * NUM_HEADS;         // 87616
    const int nBlocks = (totalWaves + 7) / 8;           // 8 waves / block
    deform_attn_kernel<<<dim3(nBlocks), blk, 0, stream>>>(
        q_buf, k_buf, off_buf, attn_buf, Bn, Np, Hh, Ww);

    // final projection: out = attn @ Wout + bout
    gemm_bf16_wmma<<<dim3(gM, C / BN), blk, 0, stream>>>(
        attn_buf, C, Wout, C, bout, out, C, Mtot, C, C);
}
